// TransBlock_5188320494455
// MI455X (gfx1250) — matrix operand order repaired and checked by reading, not compiled, not run on hardware
//
#include <hip/hip_runtime.h>
#include <hip/hip_bf16.h>

typedef _Float16 half_t;
typedef __attribute__((ext_vector_type(16))) _Float16 v16h;
typedef __attribute__((ext_vector_type(8)))  _Float16 v8h;
typedef __attribute__((ext_vector_type(4)))  _Float16 v4h;
typedef __attribute__((ext_vector_type(8)))  float    v8f;

#define NEG_SLOPE 0.2f

__device__ __forceinline__ float leaky(float x) { return x >= 0.f ? x : NEG_SLOPE * x; }

__device__ __forceinline__ v8f wmma16(v16h a, v16h b, v8f c) {
    // D = A(16x32 f16) * B(32x16 f16) + C(16x16 f32)
    return __builtin_amdgcn_wmma_f32_16x16x32_f16(false, a, false, b, (short)0, c, false, false);
}

// A fragment from row-major f16 tile (row stride 64 halves). Per ISA layout the
// two 8-half groups a lane needs are contiguous -> two ds_load_b128.
__device__ __forceinline__ v16h load_A(const half_t* tile, int lane, int kbase) {
    int m  = lane & 15;
    int ko = ((lane & 16) >> 1) + kbase;          // hi*8 + kbase
    const half_t* p = tile + m * 64 + ko;
    v8h lo = *(const v8h*)(p);
    v8h hi = *(const v8h*)(p + 16);
    return __builtin_shufflevector(lo, hi, 0,1,2,3,4,5,6,7,8,9,10,11,12,13,14,15);
}

// Stage a 64x64 f32 weight (row-major K x N) into LDS pre-swizzled into
// B-fragment order: wswz[((nt*2+ks)*32 + lane)*16 + e] = w[k*64+n],
// k = ks*32 + ((lane&16)>>1) + (e&7) + ((e>>3)<<4), n = nt*16 + (lane&15).
__device__ __forceinline__ void stage_B(const float* __restrict__ w, half_t* wswz, int tid) {
    for (int i = tid; i < 4096; i += 256) {
        int e    = i & 15;
        int lane = (i >> 4) & 31;
        int g    = i >> 9;                 // nt*2+ks
        int ks   = g & 1, nt = g >> 1;
        int k = ks * 32 + ((lane & 16) >> 1) + (e & 7) + ((e >> 3) << 4);
        int n = nt * 16 + (lane & 15);
        wswz[i] = (half_t)w[k * 64 + n];
    }
}
// One contiguous 32B read per B fragment (g = nt*2 + ks)
__device__ __forceinline__ v16h load_Bs(const half_t* wswz, int lane, int g) {
    return *(const v16h*)(wswz + (g * 32 + lane) * 16);
}

// ---------------- K0: NCHW -> (B*N, C) channels-last pack (float4 writes) ---
__global__ __launch_bounds__(256) void k_pack(const float* __restrict__ x, float* __restrict__ xbuf) {
    long idx = (long)blockIdx.x * 256 + threadIdx.x;    // over B*N*C/4
    int  c0 = (int)((idx << 2) & 63);
    long bn = idx >> 4;
    int  b  = (int)(bn >> 14);
    long n  = bn & 16383;
    const float* src = x + ((((long)b) * 64) << 14) + n;
    float4 t;
    t.x = src[((long)(c0 + 0)) << 14];
    t.y = src[((long)(c0 + 1)) << 14];
    t.z = src[((long)(c0 + 2)) << 14];
    t.w = src[((long)(c0 + 3)) << 14];
    ((float4*)xbuf)[idx] = t;
}

__global__ __launch_bounds__(256) void k_zero(float* p, int n) {
    int i = blockIdx.x * 256 + threadIdx.x;
    if (i < n) p[i] = 0.f;
}

// ---------------- K1: q,k,v = x @ W (WMMA) + sum-of-squares -----------------
__global__ __launch_bounds__(256) void k_qkv(const float* __restrict__ xbuf,
        const float* __restrict__ wq, const float* __restrict__ wk, const float* __restrict__ wv,
        half_t* __restrict__ qh, half_t* __restrict__ kh, half_t* __restrict__ vh,
        float* __restrict__ qss, float* __restrict__ kss) {
    __shared__ __align__(32) half_t sW[3][64 * 64];   // 24 KB, B-swizzled
    __shared__ __align__(32) half_t sX[128 * 64];     // 16 KB
    int tid = threadIdx.x;
    long gbase = (long)blockIdx.x * 128;
    stage_B(wq, sW[0], tid);
    stage_B(wk, sW[1], tid);
    stage_B(wv, sW[2], tid);
    const float4* xv = (const float4*)(xbuf + gbase * 64);
    for (int i = tid; i < 2048; i += 256) {
        float4 t = xv[i];
        v4h h; h[0] = (half_t)t.x; h[1] = (half_t)t.y; h[2] = (half_t)t.z; h[3] = (half_t)t.w;
        *(v4h*)(sX + i * 4) = h;
    }
    __builtin_prefetch(xbuf + (gbase + 128) * 64, 0, 1);   // next tile -> global_prefetch
    __syncthreads();

    int lane = tid & 31, w = tid >> 5;
    const half_t* at = &sX[w * 16 * 64];
    v16h a0 = load_A(at, lane, 0);
    v16h a1 = load_A(at, lane, 32);
    int  b  = (int)(gbase >> 14);
    long wtile = gbase + (long)w * 16;
    int  rofs = (lane >> 4) << 3;

#pragma unroll
    for (int mm = 0; mm < 3; ++mm) {
        half_t* op = (mm == 0) ? qh : (mm == 1) ? kh : vh;
#pragma unroll
        for (int nt = 0; nt < 4; ++nt) {
            v8f acc = {};
            acc = wmma16(a0, load_Bs(sW[mm], lane, nt * 2 + 0), acc);
            acc = wmma16(a1, load_Bs(sW[mm], lane, nt * 2 + 1), acc);
            int ch = nt * 16 + (lane & 15);
            float ss = 0.f;
#pragma unroll
            for (int r = 0; r < 8; ++r) {
                long tok = wtile + r + rofs;
                op[tok * 64 + ch] = (half_t)acc[r];
                ss += acc[r] * acc[r];
            }
            if (mm == 0)      atomicAdd(&qss[b * 64 + ch], ss);
            else if (mm == 1) atomicAdd(&kss[b * 64 + ch], ss);
        }
    }
}

// ---------------- K2: per-(b,h) 8x8 logits + softmax (VALU) -----------------
__global__ __launch_bounds__(256) void k_attn(const half_t* __restrict__ qh,
        const half_t* __restrict__ kh, const float* __restrict__ qss,
        const float* __restrict__ kss, float* __restrict__ attn) {
    int b = blockIdx.x >> 3, h = blockIdx.x & 7;
    int tid = threadIdx.x, lane = tid & 31, w = tid >> 5;
    float acc[8][8];
#pragma unroll
    for (int d = 0; d < 8; ++d)
#pragma unroll
        for (int e = 0; e < 8; ++e) acc[d][e] = 0.f;

    const half_t* qp = qh + (((long)b) << 14) * 64 + h * 8;
    const half_t* kp = kh + (((long)b) << 14) * 64 + h * 8;
    for (int n = tid; n < 16384; n += 256) {
        v8h qv = *(const v8h*)(qp + (long)n * 64);     // one global_load_b128
        v8h kv = *(const v8h*)(kp + (long)n * 64);
#pragma unroll
        for (int d = 0; d < 8; ++d)
#pragma unroll
            for (int e = 0; e < 8; ++e) acc[d][e] += (float)kv[d] * (float)qv[e];
    }

    __shared__ float sRed[8][64];
#pragma unroll
    for (int d = 0; d < 8; ++d)
#pragma unroll
        for (int e = 0; e < 8; ++e) {
            float v = acc[d][e];
            for (int off = 16; off > 0; off >>= 1) v += __shfl_down(v, off, 32);
            if (lane == 0) sRed[w][d * 8 + e] = v;
        }
    __syncthreads();
    if (tid < 8) {
        int d = tid;
        float row[8];
#pragma unroll
        for (int e = 0; e < 8; ++e) {
            float t = 0.f;
            for (int ww = 0; ww < 8; ++ww) t += sRed[ww][d * 8 + e];
            row[e] = t;
        }
        float nk = fmaxf(sqrtf(kss[b * 64 + h * 8 + d]), 1e-12f);
#pragma unroll
        for (int e = 0; e < 8; ++e)
            row[e] /= (nk * fmaxf(sqrtf(qss[b * 64 + h * 8 + e]), 1e-12f));
        float m = row[0];
#pragma unroll
        for (int e = 1; e < 8; ++e) m = fmaxf(m, row[e]);
        float s = 0.f;
#pragma unroll
        for (int e = 0; e < 8; ++e) { row[e] = expf(row[e] - m); s += row[e]; }
        float inv = 1.f / s;
#pragma unroll
        for (int e = 0; e < 8; ++e)
            attn[b * 512 + h * 64 + d * 8 + e] = row[e] * inv;
    }
}

// o element: o[tok][c] = sum_e attn[c][e] * v[tok][(c&~7)+e]  (v8h vector read)
__device__ __forceinline__ half_t o_elem(const half_t* vt, const float* sA, int m, int c) {
    const float* ar = &sA[c * 8];
    v8h vv = *(const v8h*)(vt + m * 64 + (c & ~7));
    float s = 0.f;
#pragma unroll
    for (int e2 = 0; e2 < 8; ++e2) s += ar[e2] * (float)vv[e2];
    return (half_t)s;
}

// ---------------- K3: o@Wproj (WMMA) * depthwise-gate + res + LN + FF (WMMA)
__global__ __launch_bounds__(256) void k_attnout(
        const float* __restrict__ xin, float* __restrict__ xout,
        const half_t* __restrict__ vh, const float* __restrict__ attn,
        const float* __restrict__ wproj, const float* __restrict__ bproj,
        const float* __restrict__ wpos, const float* __restrict__ ln_g,
        const float* __restrict__ ln_b, const float* __restrict__ wff) {
    __shared__ __align__(32) half_t sV[128 * 64];     // v row tile, reused for LN out
    __shared__ __align__(32) half_t sP[128 * 64];     // out_p, then x1 in-place
    __shared__ __align__(32) half_t sWp[64 * 64];     // B-swizzled
    __shared__ __align__(32) half_t sWf[64 * 64];     // B-swizzled
    __shared__ float  sAttn[512];
    __shared__ float  sB[64], sLnG[64], sLnB[64];
    __shared__ float  sWpos[576];

    int tid = threadIdx.x;
    long gbase = (long)blockIdx.x * 128;
    int  b = (int)(gbase >> 14);
    stage_B(wproj, sWp, tid);
    stage_B(wff,   sWf, tid);
    for (int i = tid; i < 1024; i += 256) *(v8h*)(sV + i * 8) = *(const v8h*)(vh + gbase * 64 + i * 8);
    for (int i = tid; i < 512;  i += 256) sAttn[i] = attn[b * 512 + i];
    for (int i = tid; i < 576;  i += 256) sWpos[i] = wpos[i];
    if (tid < 64) { sB[tid] = bproj[tid]; sLnG[tid] = ln_g[tid]; sLnB[tid] = ln_b[tid]; }
    __syncthreads();

    // ---- depthwise out_p, cooperative: tile is one image row y ----
    {
        int tok = tid >> 1, cb = (tid & 1) * 32;
        int y = (int)((gbase & 16383) >> 7);
        int x = tok;
        const half_t* vimg = vh + ((((long)b) << 14) * 64);
#pragma unroll
        for (int grp = 0; grp < 4; ++grp) {
            int ch0 = cb + grp * 8;
            float acc8[8];
#pragma unroll
            for (int j = 0; j < 8; ++j) acc8[j] = 0.f;
            for (int dy = -1; dy <= 1; ++dy) {
                int yy = y + dy;
                if (yy < 0 || yy > 127) continue;
                for (int dx = -1; dx <= 1; ++dx) {
                    int xx = x + dx;
                    if (xx < 0 || xx > 127) continue;
                    v8h vv = (dy == 0)
                        ? *(const v8h*)(sV + xx * 64 + ch0)
                        : *(const v8h*)(vimg + ((long)(yy * 128 + xx)) * 64 + ch0);
#pragma unroll
                    for (int j = 0; j < 8; ++j)
                        acc8[j] += sWpos[(ch0 + j) * 9 + (dy + 1) * 3 + (dx + 1)] * (float)vv[j];
                }
            }
            v8h hp;
#pragma unroll
            for (int j = 0; j < 8; ++j) hp[j] = (half_t)acc8[j];
            *(v8h*)(sP + tok * 64 + ch0) = hp;
        }
    }

    int lane = tid & 31, w = tid >> 5;
    int hi = (lane >> 4) & 1;
    int m = lane & 15;
    int rofs = hi << 3;
    const half_t* vt = &sV[w * 16 * 64];

    // A fragments of o (attn x v), built before sV is overwritten
    v16h oa0, oa1;
#pragma unroll
    for (int e = 0; e < 16; ++e) {
        int k0 = (e & 7) + ((e >> 3) << 4) + (hi << 3);
        oa0[e] = o_elem(vt, sAttn, m, k0);
        oa1[e] = o_elem(vt, sAttn, m, k0 + 32);
    }
    __syncthreads();   // out_p complete, o fragments captured

    float x1r[4][8];   // x1 kept in registers for the FF accumulator
#pragma unroll
    for (int nt = 0; nt < 4; ++nt) {
        v8f acc = {};
        acc = wmma16(oa0, load_Bs(sWp, lane, nt * 2 + 0), acc);
        acc = wmma16(oa1, load_Bs(sWp, lane, nt * 2 + 1), acc);
        int ch = nt * 16 + (lane & 15);
#pragma unroll
        for (int r = 0; r < 8; ++r) {
            int  trow = w * 16 + r + rofs;
            long tokg = gbase + trow;
            float p  = (float)sP[trow * 64 + ch];
            float x1 = (acc[r] + sB[ch]) * p + xin[tokg * 64 + ch];
            x1r[nt][r] = x1;
            sP[trow * 64 + ch] = (half_t)x1;      // in-place: same slot, same lane
        }
    }
    __syncthreads();

    // LayerNorm over C per token -> sV (reuse)
    if (tid < 128) {
        float mean = 0.f;
        for (int c = 0; c < 64; ++c) mean += (float)sP[tid * 64 + c];
        mean *= (1.f / 64.f);
        float var = 0.f;
        for (int c = 0; c < 64; ++c) {
            float d = (float)sP[tid * 64 + c] - mean;
            var += d * d;
        }
        var *= (1.f / 64.f);
        float inv = rsqrtf(var + 1e-5f);
        for (int c = 0; c < 64; ++c) {
            float d = ((float)sP[tid * 64 + c] - mean) * inv;
            sV[tid * 64 + c] = (half_t)(d * sLnG[c] + sLnB[c]);
        }
    }
    __syncthreads();

    // x2 = LN(x1) @ wff + x1
    const half_t* lt = &sV[w * 16 * 64];
    v16h la0 = load_A(lt, lane, 0);
    v16h la1 = load_A(lt, lane, 32);
#pragma unroll
    for (int nt = 0; nt < 4; ++nt) {
        int ch = nt * 16 + (lane & 15);
        v8f acc;
#pragma unroll
        for (int r = 0; r < 8; ++r) acc[r] = x1r[nt][r];
        acc = wmma16(la0, load_Bs(sWf, lane, nt * 2 + 0), acc);
        acc = wmma16(la1, load_Bs(sWf, lane, nt * 2 + 1), acc);
#pragma unroll
        for (int r = 0; r < 8; ++r)
            xout[(gbase + (long)w * 16 + r + rofs) * 64 + ch] = acc[r];
    }
}

// ---------------- K4: per-channel stats of leaky(x) (float4) ---------------
__global__ __launch_bounds__(256) void k_bnstats(const float* __restrict__ xbuf,
        float* __restrict__ chsum, float* __restrict__ chsumsq) {
    long base = (long)blockIdx.x * 65536;
    int tid = threadIdx.x;
    const float4* p = (const float4*)(xbuf + base);
    float s[4] = {0.f, 0.f, 0.f, 0.f}, q[4] = {0.f, 0.f, 0.f, 0.f};
    for (int i = 0; i < 64; ++i) {
        float4 v = p[tid + i * 256];
        float a0 = leaky(v.x), a1 = leaky(v.y), a2 = leaky(v.z), a3 = leaky(v.w);
        s[0] += a0; q[0] += a0 * a0;
        s[1] += a1; q[1] += a1 * a1;
        s[2] += a2; q[2] += a2 * a2;
        s[3] += a3; q[3] += a3 * a3;
    }
    __shared__ float sS[1024], sQ[1024];
#pragma unroll
    for (int j = 0; j < 4; ++j) { sS[tid * 4 + j] = s[j]; sQ[tid * 4 + j] = q[j]; }
    __syncthreads();
    if (tid < 64) {
        int c = tid, j = c & 3, t0 = (c >> 2) & 15;
        float ts = 0.f, tq = 0.f;
        for (int t = 0; t < 16; ++t) {
            int th = t0 + 16 * t;
            ts += sS[th * 4 + j]; tq += sQ[th * 4 + j];
        }
        atomicAdd(&chsum[c], ts);
        atomicAdd(&chsumsq[c], tq);
    }
}

// ---------------- K5: BN1(leaky(x)) @ wd (WMMA), leaky, BN2 stats, -> NCHW --
__global__ __launch_bounds__(256) void k_conv1x1(const float* __restrict__ xbuf,
        const float* __restrict__ wd, const float* __restrict__ chsum,
        const float* __restrict__ chsumsq, const float* __restrict__ bn1g,
        const float* __restrict__ bn1b, float* __restrict__ uout,
        float* __restrict__ ch2sum, float* __restrict__ ch2sumsq) {
    __shared__ __align__(32) half_t sWd[4096];        // B-swizzled
    __shared__ __align__(32) half_t sX[128 * 64];
    __shared__ float  sScale[64], sShift[64], sRs[64], sRq[64];
    int tid = threadIdx.x;
    if (tid < 64) {
        const float M = 262144.f;
        float mean = chsum[tid] / M;
        float var  = chsumsq[tid] / M - mean * mean;
        float sc   = bn1g[tid] * rsqrtf(var + 1e-4f);
        sScale[tid] = sc;
        sShift[tid] = bn1b[tid] - mean * sc;
        sRs[tid] = 0.f; sRq[tid] = 0.f;
    }
    stage_B(wd, sWd, tid);
    __syncthreads();
    long gbase = (long)blockIdx.x * 128;
    const float4* xv = (const float4*)(xbuf + gbase * 64);
    for (int i = tid; i < 2048; i += 256) {
        float4 t = xv[i];
        int ch = (i * 4) & 63;
        v4h h;
        h[0] = (half_t)(sScale[ch + 0] * leaky(t.x) + sShift[ch + 0]);
        h[1] = (half_t)(sScale[ch + 1] * leaky(t.y) + sShift[ch + 1]);
        h[2] = (half_t)(sScale[ch + 2] * leaky(t.z) + sShift[ch + 2]);
        h[3] = (half_t)(sScale[ch + 3] * leaky(t.w) + sShift[ch + 3]);
        *(v4h*)(sX + i * 4) = h;
    }
    __syncthreads();

    int lane = tid & 31, w = tid >> 5;
    const half_t* at = &sX[w * 16 * 64];
    v16h a0 = load_A(at, lane, 0);
    v16h a1 = load_A(at, lane, 32);
    int  b = (int)(gbase >> 14);
    int  rofs = (lane >> 4) << 3;
#pragma unroll
    for (int nt = 0; nt < 4; ++nt) {
        v8f acc = {};
        acc = wmma16(a0, load_Bs(sWd, lane, nt * 2 + 0), acc);
        acc = wmma16(a1, load_Bs(sWd, lane, nt * 2 + 1), acc);
        int ch = nt * 16 + (lane & 15);
        float ls = 0.f, lq = 0.f;
#pragma unroll
        for (int r = 0; r < 8; ++r) {
            float u = leaky(acc[r]);
            long tok = (gbase & 16383) + (long)w * 16 + r + rofs;
            uout[(((long)(b * 64 + ch)) << 14) + tok] = u;   // NCHW
            ls += u; lq += u * u;
        }
        atomicAdd(&sRs[ch], ls);
        atomicAdd(&sRq[ch], lq);
    }
    __syncthreads();
    if (tid < 64) {
        atomicAdd(&ch2sum[tid],  sRs[tid]);
        atomicAdd(&ch2sumsq[tid], sRq[tid]);
    }
}

// ---------------- K6: z = BN2(u) + x_orig; 8x8 pool + per-(b,c) mean --------
__global__ __launch_bounds__(256) void k_residual_pool(float* __restrict__ z,
        const float* __restrict__ xin, const float* __restrict__ ch2sum,
        const float* __restrict__ ch2sumsq, const float* __restrict__ bn2g,
        const float* __restrict__ bn2b, float* __restrict__ pool,
        float* __restrict__ zsum) {
    int bc = blockIdx.x;
    int c  = bc & 63;
    int tid = threadIdx.x;
    __shared__ float sPool[64];
    if (tid < 64) sPool[tid] = 0.f;
    const float M = 262144.f;
    float mean = ch2sum[c] / M;
    float var  = ch2sumsq[c] / M - mean * mean;
    float sc   = bn2g[c] * rsqrtf(var + 1e-4f);
    float sh   = bn2b[c] - mean * sc;
    __syncthreads();
    long base = ((long)bc) << 14;
    float4* zp = (float4*)(z + base);
    const float4* xp = (const float4*)(xin + base);
    for (int i = tid; i < 4096; i += 256) {
        float4 u = zp[i], xr = xp[i], zz;
        zz.x = sc * u.x + sh + xr.x;
        zz.y = sc * u.y + sh + xr.y;
        zz.z = sc * u.z + sh + xr.z;
        zz.w = sc * u.w + sh + xr.w;
        zp[i] = zz;
        int n = i * 4, y = n >> 7, xq = n & 127;
        atomicAdd(&sPool[(y >> 4) * 8 + (xq >> 4)], zz.x + zz.y + zz.z + zz.w);
    }
    __syncthreads();
    if (tid < 64) pool[(long)bc * 64 + tid] = sPool[tid] * (1.f / 256.f);
    if (tid == 0) {
        float t = 0.f;
        for (int i = 0; i < 64; ++i) t += sPool[i];
        zsum[bc] = t * (1.f / 16384.f);
    }
}

// ---------------- K7: SAA gate (min/max, correlation, SE-MLP) — 1 block -----
__global__ __launch_bounds__(256) void k_saa_gate(const float* __restrict__ pool,
        const float* __restrict__ zsum, const float* __restrict__ w1,
        const float* __restrict__ w2, float* __restrict__ gvec) {
    int tid = threadIdx.x;
    __shared__ float sMin[256], sMax[256];
    float mn = 3.4e38f, mx = -3.4e38f;
    for (int i = tid; i < 65536; i += 256) {
        float v = pool[i];
        mn = fminf(mn, v); mx = fmaxf(mx, v);
    }
    sMin[tid] = mn; sMax[tid] = mx;
    __syncthreads();
    for (int off = 128; off > 0; off >>= 1) {
        if (tid < off) {
            sMin[tid] = fminf(sMin[tid], sMin[tid + off]);
            sMax[tid] = fmaxf(sMax[tid], sMax[tid + off]);
        }
        __syncthreads();
    }
    float gmn = sMin[0];
    float inv = 1.f / (sMax[0] - gmn);
    __syncthreads();

    __shared__ float sS[1024];
    for (int idx = tid; idx < 1024; idx += 256) {
        int b = idx >> 6, i = idx & 63;
        float t = 0.f;
        for (int c = 0; c < 64; ++c) t += (pool[((b * 64 + c) * 64) + i] - gmn);
        sS[idx] = t * inv;
    }
    __syncthreads();
    __shared__ float sG[1024];
    for (int idx = tid; idx < 1024; idx += 256) {
        int b = idx >> 6, c = idx & 63;
        float t = 0.f;
        for (int i = 0; i < 64; ++i)
            t += (pool[((b * 64 + c) * 64) + i] - gmn) * inv * sS[b * 64 + i];
        sG[idx] = (t * (1.f / 64.f)) * zsum[idx];
    }
    __syncthreads();
    __shared__ float sH[64];
    if (tid < 64) {
        int b = tid >> 2, j = tid & 3;
        float t = 0.f;
        for (int c = 0; c < 64; ++c) t += sG[b * 64 + c] * w1[c * 4 + j];
        sH[tid] = leaky(t);
    }
    __syncthreads();
    for (int idx = tid; idx < 1024; idx += 256) {
        int b = idx >> 6, c = idx & 63;
        float t = 0.f;
        for (int j = 0; j < 4; ++j) t += sH[b * 4 + j] * w2[j * 64 + c];
        gvec[idx] = 1.f / (1.f + expf(-t));
    }
}

// ---------------- K8: out *= g[b,c] (NCHW, float4 in place) -----------------
__global__ __launch_bounds__(256) void k_scale_out(float* __restrict__ out,
        const float* __restrict__ gvec) {
    long idx = (long)blockIdx.x * 256 + threadIdx.x;    // over B*C*H*W/4
    float g = gvec[(int)(idx >> 12)];
    float4* o = (float4*)out;
    float4 v = o[idx];
    v.x *= g; v.y *= g; v.z *= g; v.w *= g;
    o[idx] = v;
}

extern "C" void kernel_launch(void* const* d_in, const int* in_sizes, int n_in,
                              void* d_out, int out_size, void* d_ws, size_t ws_size,
                              hipStream_t stream) {
    (void)in_sizes; (void)n_in; (void)out_size; (void)ws_size;
    const float* x     = (const float*)d_in[0];
    const float* wq    = (const float*)d_in[1];
    const float* wk    = (const float*)d_in[2];
    const float* wv    = (const float*)d_in[3];
    const float* wproj = (const float*)d_in[4];
    const float* bproj = (const float*)d_in[5];
    const float* wpos  = (const float*)d_in[6];
    const float* ln_g  = (const float*)d_in[7];
    const float* ln_b  = (const float*)d_in[8];
    const float* wff   = (const float*)d_in[9];
    const float* bn1g  = (const float*)d_in[10];
    const float* bn1b  = (const float*)d_in[11];
    const float* wd    = (const float*)d_in[12];
    const float* bn2g  = (const float*)d_in[13];
    const float* bn2b  = (const float*)d_in[14];
    const float* saa1  = (const float*)d_in[15];
    const float* saa2  = (const float*)d_in[16];
    float* out = (float*)d_out;

    char* ws = (char*)d_ws;
    float*  xbuf = (float*)ws;                              // 67 MB  (B*N, C) f32
    half_t* qh   = (half_t*)(ws + 67108864);
    half_t* kh   = (half_t*)(ws + 100663296);
    half_t* vh   = (half_t*)(ws + 134217728);
    float*  st   = (float*)(ws + 167772160);
    float* qss   = st;            // 1024
    float* kss   = st + 1024;     // 1024
    float* attnb = st + 2048;     // 8192
    float* chsum   = st + 10240;  // 4 x 64 contiguous
    float* chsumsq = chsum + 64;
    float* ch2sum  = chsum + 128;
    float* ch2sumsq= chsum + 192;
    float* pool  = chsum + 256;   // 65536
    float* zsum  = pool + 65536;  // 1024
    float* gvec  = zsum + 1024;   // 1024

    k_pack<<<16384, 256, 0, stream>>>(x, xbuf);

    for (int l = 0; l < 2; ++l) {
        k_zero<<<8, 256, 0, stream>>>(qss, 2048);
        k_qkv<<<2048, 256, 0, stream>>>(xbuf, wq + l * 4096, wk + l * 4096,
                                        wv + l * 4096, qh, kh, vh, qss, kss);
        k_attn<<<128, 256, 0, stream>>>(qh, kh, qss, kss, attnb);
        k_attnout<<<2048, 256, 0, stream>>>(xbuf, xbuf, vh, attnb,
                wproj + l * 4096, bproj + l * 64, wpos + l * 576,
                ln_g + l * 64, ln_b + l * 64, wff + l * 4096);
    }

    k_zero<<<1, 256, 0, stream>>>(chsum, 256);
    k_bnstats<<<256, 256, 0, stream>>>(xbuf, chsum, chsumsq);
    k_conv1x1<<<2048, 256, 0, stream>>>(xbuf, wd, chsum, chsumsq, bn1g, bn1b,
                                        out, ch2sum, ch2sumsq);
    k_residual_pool<<<1024, 256, 0, stream>>>(out, x, ch2sum, ch2sumsq,
                                              bn2g, bn2b, pool, zsum);
    k_saa_gate<<<1, 256, 0, stream>>>(pool, zsum, saa1, saa2, gvec);
    k_scale_out<<<16384, 256, 0, stream>>>(out, gvec);
}
